// DRAGON_34479997453011
// MI455X (gfx1250) — matrix operand order repaired and checked by reading, not compiled, not run on hardware
//
#include <hip/hip_runtime.h>
#include <math.h>

// ---------------- problem constants (match reference) ----------------
static constexpr int  kNU = 20000;
static constexpr int  kNI = 8000;
static constexpr int  kE  = 64;          // embedding dim per modality
static constexpr int  kD2 = 2 * kE;      // fused dim = 128
static constexpr int  kDV = 4096;
static constexpr int  kDT = 768;
static constexpr int  kNE = 200000;
static constexpr int  kB  = 4096;
static constexpr int  kKI = 10;          // K_ISG
static constexpr int  kKU = 20;          // K_UCG
static constexpr int  kN  = kNU + kNI;   // 28000
static constexpr float kMMW = 0.1f;
static constexpr float kWD  = 1e-4f;

// ---------------- WMMA types / helpers ----------------
typedef __attribute__((ext_vector_type(16))) __bf16    v16bf;
typedef __attribute__((ext_vector_type(8)))  float     v8f;
typedef __attribute__((ext_vector_type(4)))  unsigned  v4u;
typedef __attribute__((ext_vector_type(8)))  int       v8i;
typedef __attribute__((ext_vector_type(4)))  int       v4i;

union V16Q { v4u q[2]; v16bf v; };

__device__ __forceinline__ unsigned bf16pair(float lo, float hi) {
  unsigned a = __builtin_bit_cast(unsigned, lo);
  unsigned b = __builtin_bit_cast(unsigned, hi);
  a = (a + 0x7FFFu + ((a >> 16) & 1u)) >> 16;   // round-to-nearest-even
  b = (b + 0x7FFFu + ((b >> 16) & 1u)) >> 16;
  return a | (b << 16);
}

// ---------------- Tensor Data Mover: contiguous global -> LDS ----------------
#if defined(__has_builtin)
#if __has_builtin(__builtin_amdgcn_tensor_load_to_lds) && __has_builtin(__builtin_amdgcn_s_wait_tensorcnt)
#define HAVE_TDM 1
#endif
#endif

// Require the TDM path in the device pass: fail loudly if the builtin is absent
// so the compile result tells us whether tensor ops are actually emitted.
#if defined(__HIP_DEVICE_COMPILE__) && !defined(HAVE_TDM)
#error "gfx1250 tensor_load_to_lds builtin not available on this toolchain"
#endif

// 1-D TDM copy of n8*8 bytes from gptr to LDS byte offset ldsByteOff.
// D# layout per CDNA5 ISA §8.3/8.4: group0 {count,lds_addr,global_addr,type=2},
// group1 {data_size=3 (8B), tensor_dim0=tile_dim0=n8, tensor_dim1=1, stride0=n8}.
__device__ __forceinline__ void tdm_copy_to_lds(unsigned ldsByteOff, const void* gptr,
                                                unsigned n8) {
#ifdef HAVE_TDM
  unsigned long long ga = (unsigned long long)gptr;
  v4u g0;
  g0.x = 1u;                                            // count=1, user mode
  g0.y = ldsByteOff;                                    // lds_addr
  g0.z = (unsigned)(ga & 0xFFFFFFFFu);                  // global_addr[31:0]
  g0.w = (unsigned)((ga >> 32) & 0x01FFFFFFu) | (2u << 30);  // addr[56:32] | type=2
  v8i g1;
  g1[0] = (int)(3u << 16);                              // data_size = 8 bytes
  g1[1] = (int)((n8 & 0xFFFFu) << 16);                  // tensor_dim0[15:0]
  g1[2] = (int)(((n8 >> 16) & 0xFFFFu) | (1u << 16));   // tensor_dim0[31:16] | tensor_dim1=1
  g1[3] = (int)((n8 & 0xFFFFu) << 16);                  // tile_dim0
  g1[4] = 0;                                            // tile_dim1/2 unused
  g1[5] = (int)n8;                                      // tensor_dim0_stride[31:0]
  g1[6] = 0;
  g1[7] = 0;
  v4i z4 = {0, 0, 0, 0};
#if __clang_major__ >= 23
  v8i z8 = {0, 0, 0, 0, 0, 0, 0, 0};
  __builtin_amdgcn_tensor_load_to_lds(g0, g1, z4, z4, z8, 0);
#else
  __builtin_amdgcn_tensor_load_to_lds(g0, g1, z4, z4, 0);
#endif
  __builtin_amdgcn_s_wait_tensorcnt((short)0);
#endif
}

// ---------------- trivial utility kernels ----------------
__global__ void zero_f32(float* p, long n) {
  long t = (long)blockIdx.x * blockDim.x + threadIdx.x;
  long s = (long)gridDim.x * blockDim.x;
  for (long i = t; i < n; i += s) p[i] = 0.0f;
}
__global__ void copy_f32(float* dst, const float* src, long n) {
  long t = (long)blockIdx.x * blockDim.x + threadIdx.x;
  long s = (long)gridDim.x * blockDim.x;
  for (long i = t; i < n; i += s) dst[i] = src[i];
}
__global__ void add_f32(float* dst, const float* src, long n) {
  long t = (long)blockIdx.x * blockDim.x + threadIdx.x;
  long s = (long)gridDim.x * blockDim.x;
  for (long i = t; i < n; i += s) dst[i] += src[i];
}

// ---------------- degree ----------------
__global__ void degree_kernel(const int* __restrict__ eu, const int* __restrict__ ei,
                              float* __restrict__ deg) {
  int e = blockIdx.x * blockDim.x + threadIdx.x;
  if (e >= kNE) return;
  atomicAdd(&deg[eu[e]], 1.0f);
  atomicAdd(&deg[kNU + ei[e]], 1.0f);
}

// ---------------- per-row L2 norm: invn = 1/||r||, nrm = ||r|| ----------------
__global__ void rownorm(const float* __restrict__ feat, float* __restrict__ invn,
                        float* __restrict__ nrm, int rows, int Kdim) {
  const int wave = (int)(((long)blockIdx.x * blockDim.x + threadIdx.x) >> 5);
  const int lane = threadIdx.x & 31;
  if (wave >= rows) return;
  const float* r = feat + (long)wave * Kdim;
  float s = 0.0f;
  for (int k = lane; k < Kdim; k += 32) { const float v = r[k]; s += v * v; }
#pragma unroll
  for (int o = 16; o; o >>= 1) s += __shfl_down(s, o, 32);
  if (lane == 0) {
    const float ss = fmaxf(s, 1e-30f);
    invn[wave] = rsqrtf(ss);
    nrm[wave]  = sqrtf(ss);
  }
}

// --------- pack normalized rows to bf16 pairs, ascending-K order -----------
// One canonical layout feeds BOTH WMMA operands with contiguous vector loads:
//   B-lane (N=r%16, half h):  dwords [kb*16 + 8h .. +7]        (32B load)
//   A-lane (M=r%16, half h):  dwords [kb*16+8h..+3],[kb*16+8h+4..+3]  (2x16B)
__global__ void pack_bf16(const float* __restrict__ feat, const float* __restrict__ invn,
                          unsigned* __restrict__ outp, int Kdim, long total) {
  long t = (long)blockIdx.x * blockDim.x + threadIdx.x;
  long s = (long)gridDim.x * blockDim.x;
  const int rd = Kdim / 2;
  for (long i = t; i < total; i += s) {
    const long r = i / rd;
    const int  d = (int)(i - r * rd);
    const float sc = invn[r];
    const float* f = feat + r * (long)Kdim + 2 * d;
    outp[i] = bf16pair(f[0] * sc, f[1] * sc);
  }
}

// ---- pack W[K,64] column-major to bf16 pairs: Wp[c][K/2] dwords ----
__global__ void pack_w(const float* __restrict__ W, unsigned* __restrict__ Wp,
                       int Kdim) {
  long t = (long)blockIdx.x * blockDim.x + threadIdx.x;
  long s = (long)gridDim.x * blockDim.x;
  const int rd = Kdim / 2;
  const long total = (long)kE * rd;
  for (long i = t; i < total; i += s) {
    const long c = i / rd;
    const int  d = (int)(i - c * rd);
    Wp[i] = bf16pair(W[(long)(2 * d) * kE + c], W[(long)(2 * d + 1) * kE + c]);
  }
}

// ------- fused Gram (fn @ fn^T, bf16 WMMA / f32 acc) + per-row top-10 -------
// One wave per 16-row block. A panel (16 rows, packed bf16) staged into LDS via
// the Tensor Data Mover (single contiguous region); B streamed from L2.
template <int KD>
__global__ void topk_sim(const unsigned* __restrict__ fnp, int* __restrict__ knn) {
  constexpr int RD  = KD / 2;       // dwords per row
  constexpr int NKB = KD / 32;      // K-blocks
  extern __shared__ unsigned ldsA[];                 // [16][RD] + sim[256]
  float* simLds = (float*)(ldsA + 16 * RD);
  const int lane = threadIdx.x;
  const int m = lane & 15, half = lane >> 4;
  const int i0 = blockIdx.x * 16;

  // stage the 16-row A panel (contiguous in fnp) into LDS
  const unsigned* gsrc = fnp + (long)i0 * RD;
#ifdef HAVE_TDM
  tdm_copy_to_lds(__builtin_amdgcn_groupstaticsize(), gsrc, (16u * RD * 4u) / 8u);
#else
  for (int idx = lane; idx < 16 * RD; idx += 32) ldsA[idx] = gsrc[idx];
#endif
  __syncthreads();

  float tv[kKI]; int ti[kKI];
#pragma unroll
  for (int t = 0; t < kKI; ++t) { tv[t] = -3e38f; ti[t] = 0; }

  for (int j0 = 0; j0 < kNI; j0 += 16) {
    if (j0 + 16 < kNI)
      __builtin_prefetch(fnp + (long)(j0 + 16 + m) * RD, 0, 1);   // global_prefetch_b8
    const unsigned* brow = fnp + (long)(j0 + m) * RD + 8 * half;
    const unsigned* arow = ldsA + m * RD;
    v8f c = {};
#pragma unroll 4
    for (int kb = 0; kb < NKB; ++kb) {
      V16Q a;
      a.q[0] = *(const v4u*)(arow + kb * 16 + 8 * half);
      a.q[1] = *(const v4u*)(arow + kb * 16 + 8 * half + 4);
      v16bf b = *(const v16bf*)(brow + kb * 16);
      c = __builtin_amdgcn_wmma_f32_16x16x32_bf16(false, a.v, false, b, (short)0, c,
                                                  false, false);
    }
#pragma unroll
    for (int r = 0; r < 8; ++r) simLds[(r + 8 * half) * 16 + m] = c[r];
    __syncthreads();
    if (lane < 16) {
#pragma unroll
      for (int n = 0; n < 16; ++n) {
        const float v = simLds[lane * 16 + n];
        if (v > tv[kKI - 1]) {                 // stable: ties keep earlier index
          int p = kKI - 1;
          while (p > 0 && tv[p - 1] < v) { tv[p] = tv[p - 1]; ti[p] = ti[p - 1]; --p; }
          tv[p] = v; ti[p] = j0 + n;
        }
      }
    }
    __syncthreads();
  }
  if (lane < 16)
    for (int t = 0; t < kKI; ++t) knn[(long)(i0 + lane) * kKI + t] = ti[t];
}

// --------- projection via packed operands: P = diag(nrm) * (fn @ W) ----------
// feat @ W == diag(||feat_i||) * (fn @ W), so reuse the normalized bf16 pack.
template <int KD>
__global__ void proj_wmma(const unsigned* __restrict__ fnp, const unsigned* __restrict__ Wp,
                          const float* __restrict__ nrm, float* __restrict__ xout) {
  constexpr int RD  = KD / 2;
  constexpr int NKB = KD / 32;
  const int rb = blockIdx.x;           // row block (16 rows)
  const int cb = blockIdx.y;           // col block (16 of 64 cols)
  const int lane = threadIdx.x;        // wave32
  const int m = lane & 15, half = lane >> 4;
  const unsigned* arow = fnp + (long)(rb * 16 + m) * RD;
  const unsigned* brow = Wp + (long)(cb * 16 + m) * RD + 8 * half;
  v8f c = {};
#pragma unroll 4
  for (int kb = 0; kb < NKB; ++kb) {
    V16Q a;
    a.q[0] = *(const v4u*)(arow + kb * 16 + 8 * half);
    a.q[1] = *(const v4u*)(arow + kb * 16 + 8 * half + 4);
    v16bf b = *(const v16bf*)(brow + kb * 16);
    c = __builtin_amdgcn_wmma_f32_16x16x32_bf16(false, a.v, false, b, (short)0, c,
                                                false, false);
  }
#pragma unroll
  for (int r = 0; r < 8; ++r) {
    const int rr = rb * 16 + r + 8 * half;
    xout[(long)rr * kE + cb * 16 + m] = c[r] * nrm[rr];
  }
}

// ---------------- hetero-GCN scatter: hn[dst] += h[src] * norm ----------------
__global__ void gcn_scatter(const float* __restrict__ h, float* __restrict__ hn,
                            const int* __restrict__ eu, const int* __restrict__ ei,
                            const float* __restrict__ deg) {
  long t = (long)blockIdx.x * blockDim.x + threadIdx.x;
  const long total = 2L * kNE * kE;
  if (t >= total) return;
  const int d = (int)(t & (kE - 1));
  const long e2 = t >> 6;
  int s, dd;
  if (e2 < kNE) { s = eu[e2];            dd = kNU + ei[e2]; }
  else          { long e = e2 - kNE; s = kNU + ei[e]; dd = eu[e]; }
  const float nrm = rsqrtf(fmaxf(deg[s], 1.0f) * fmaxf(deg[dd], 1.0f));
  atomicAdd(&hn[(long)dd * kE + d], h[(long)s * kE + d] * nrm);
}

// ---------------- fuse modalities ----------------
__global__ void combine_feats(const float* __restrict__ ov, const float* __restrict__ ot,
                              const float* __restrict__ alpha_p,
                              float* __restrict__ u_f, float* __restrict__ i_f) {
  long t = (long)blockIdx.x * blockDim.x + threadIdx.x;
  const long total = (long)kN * kE;
  if (t >= total) return;
  const int d = (int)(t & (kE - 1));
  const long n = t >> 6;
  const float a = 1.0f / (1.0f + expf(-alpha_p[0]));
  if (n < kNU) {
    u_f[n * kD2 + d]       = a * ov[n * kE + d];
    u_f[n * kD2 + kE + d]  = (1.0f - a) * ot[n * kE + d];
  } else {
    const long i = n - kNU;
    i_f[i * kD2 + d]       = ov[n * kE + d];
    i_f[i * kD2 + kE + d]  = ot[n * kE + d];
  }
}

// ---------------- item homogeneous hop: z_i = i_f + adj @ i_f ----------------
// adj rows always sum to 10 (K ones per modality), so weights are 0.01 / 0.09.
__global__ void item_homo(const float* __restrict__ i_f, const int* __restrict__ knv,
                          const int* __restrict__ knt, float* __restrict__ z_i) {
  long t = (long)blockIdx.x * blockDim.x + threadIdx.x;
  const long total = (long)kNI * kD2;
  if (t >= total) return;
  const int d = (int)(t & (kD2 - 1));
  const long i = t >> 7;
  float s1 = 0.0f, s2 = 0.0f;
#pragma unroll
  for (int k = 0; k < kKI; ++k) {
    s1 += i_f[(long)knv[i * kKI + k] * kD2 + d];
    s2 += i_f[(long)knt[i * kKI + k] * kD2 + d];
  }
  z_i[t] = i_f[t] + (kMMW / 10.0f) * s1 + ((1.0f - kMMW) / 10.0f) * s2;
}

// ------- user attention hop: z_u = u_f + softmax(w) @ u_f[nb] -------
__global__ void user_homo(const float* __restrict__ u_f, const int* __restrict__ nb,
                          const float* __restrict__ nbw, float* __restrict__ z_u) {
  const long u = blockIdx.x;
  const int d = threadIdx.x;           // 128 threads
  float w[kKU]; float mx = -3e38f;
#pragma unroll
  for (int k = 0; k < kKU; ++k) { w[k] = nbw[u * kKU + k]; mx = fmaxf(mx, w[k]); }
  float ssum = 0.0f;
#pragma unroll
  for (int k = 0; k < kKU; ++k) { w[k] = expf(w[k] - mx); ssum += w[k]; }
  float h = 0.0f;
#pragma unroll
  for (int k = 0; k < kKU; ++k)
    h += w[k] * u_f[(long)nb[u * kKU + k] * kD2 + d];
  z_u[u * kD2 + d] = u_f[u * kD2 + d] + h / ssum;
}

// ---------------- BPR loss sum ----------------
__global__ void bpr_kernel(const float* __restrict__ z_u, const float* __restrict__ z_i,
                           const int* __restrict__ uid, const int* __restrict__ pid,
                           const int* __restrict__ nid, float* __restrict__ acc) {
  const long t = (long)blockIdx.x * blockDim.x + threadIdx.x;
  const int b = (int)(t >> 5);
  const int lane = (int)(t & 31);
  if (b >= kB) return;
  const float* zu = z_u + (long)uid[b] * kD2;
  const float* zp = z_i + (long)pid[b] * kD2;
  const float* zn = z_i + (long)nid[b] * kD2;
  float ps = 0.0f, ns = 0.0f;
  for (int d = lane; d < kD2; d += 32) { const float u = zu[d]; ps += u * zp[d]; ns += u * zn[d]; }
#pragma unroll
  for (int o = 16; o; o >>= 1) { ps += __shfl_down(ps, o, 32); ns += __shfl_down(ns, o, 32); }
  if (lane == 0) {
    const float x = ps - ns;
    const float logsig = -(fmaxf(-x, 0.0f) + log1pf(expf(-fabsf(x))));
    atomicAdd(acc, -logsig);
  }
}

// ---------------- sum of squares (regularizer) ----------------
__global__ void sq_reduce(const float* __restrict__ a, long n, float* __restrict__ acc) {
  long t = (long)blockIdx.x * blockDim.x + threadIdx.x;
  const long s = (long)gridDim.x * blockDim.x;
  float sum = 0.0f;
  for (long i = t; i < n; i += s) { const float v = a[i]; sum += v * v; }
#pragma unroll
  for (int o = 16; o; o >>= 1) sum += __shfl_down(sum, o, 32);
  if ((threadIdx.x & 31) == 0) atomicAdd(acc, sum);
}

__global__ void finalize_kernel(const float* __restrict__ scal, const float* __restrict__ alpha_p,
                                float* __restrict__ out) {
  if (threadIdx.x == 0 && blockIdx.x == 0) {
    const float a = alpha_p[0];
    out[0] = scal[0] / (float)kB + kWD * (scal[1] + a * a) * 0.5f;
  }
}

// ---------------- driver ----------------
extern "C" void kernel_launch(void* const* d_in, const int* in_sizes, int n_in,
                              void* d_out, int out_size, void* d_ws, size_t ws_size,
                              hipStream_t stream) {
  const float* feat_v  = (const float*)d_in[0];
  const float* feat_t  = (const float*)d_in[1];
  const float* pref_v  = (const float*)d_in[2];
  const float* pref_t  = (const float*)d_in[3];
  const float* W_v     = (const float*)d_in[4];
  const float* W_t     = (const float*)d_in[5];
  const float* alpha   = (const float*)d_in[6];
  const float* nbw     = (const float*)d_in[7];
  const int*   eu      = (const int*)d_in[8];
  const int*   ei      = (const int*)d_in[9];
  const int*   u_ids   = (const int*)d_in[10];
  const int*   pos_ids = (const int*)d_in[11];
  const int*   neg_ids = (const int*)d_in[12];
  const int*   user_nb = (const int*)d_in[13];

  char* w = (char*)d_ws;
  auto alloc = [&](size_t bytes) -> void* {
    void* p = (void*)w;
    w += (bytes + 255) & ~(size_t)255;
    return p;
  };
  // ---- small persistent region ----
  float* deg    = (float*)alloc((size_t)kN * 4);
  float* invn_v = (float*)alloc((size_t)kNI * 4);
  float* invn_t = (float*)alloc((size_t)kNI * 4);
  float* nrm_v  = (float*)alloc((size_t)kNI * 4);
  float* nrm_t  = (float*)alloc((size_t)kNI * 4);
  int*   knn_v  = (int*)alloc((size_t)kNI * kKI * 4);
  int*   knn_t  = (int*)alloc((size_t)kNI * kKI * 4);
  float* proj_v = (float*)alloc((size_t)kNI * kE * 4);
  float* proj_t = (float*)alloc((size_t)kNI * kE * 4);
  unsigned* Wp_v = (unsigned*)alloc((size_t)kE * (kDV / 2) * 4);
  unsigned* Wp_t = (unsigned*)alloc((size_t)kE * (kDT / 2) * 4);
  float* scal   = (float*)alloc(256);   // [0]=bpr_sum [1]=reg_sum
  // ---- big arena, reused across phases ----
  char* arena = w;
  // phase 1 (KNN + projections): packed normalized bf16 features
  unsigned* fnv = (unsigned*)arena;                                   // NI*DV/2 dwords
  unsigned* fnt = (unsigned*)(arena + (size_t)kNI * (kDV / 2) * 4);   // NI*DT/2 dwords
  // phase 2 (GCN + fusion): overlaps phase-1 buffers (phase 1 fully done first)
  char* a2 = arena;
  auto alloc2 = [&](size_t bytes) -> void* {
    void* p = (void*)a2;
    a2 += (bytes + 255) & ~(size_t)255;
    return p;
  };
  float* hA    = (float*)alloc2((size_t)kN * kE * 4);
  float* hB    = (float*)alloc2((size_t)kN * kE * 4);
  float* out_v = (float*)alloc2((size_t)kN * kE * 4);
  float* out_t = (float*)alloc2((size_t)kN * kE * 4);
  float* u_f   = (float*)alloc2((size_t)kNU * kD2 * 4);
  float* i_f   = (float*)alloc2((size_t)kNI * kD2 * 4);
  float* z_u   = (float*)alloc2((size_t)kNU * kD2 * 4);
  float* z_i   = (float*)alloc2((size_t)kNI * kD2 * 4);

  const long nNE = (long)kN * kE;
  const int GS = 2048, BS = 256;
  const int scatterBlocks = (int)((2L * kNE * kE + BS - 1) / BS);

  // ===== phase 1: packing, projections, item-semantic KNN =====
  rownorm<<<kNI / 8, 256, 0, stream>>>(feat_v, invn_v, nrm_v, kNI, kDV);
  rownorm<<<kNI / 8, 256, 0, stream>>>(feat_t, invn_t, nrm_t, kNI, kDT);
  pack_bf16<<<GS, BS, 0, stream>>>(feat_v, invn_v, fnv, kDV, (long)kNI * (kDV / 2));
  pack_bf16<<<GS, BS, 0, stream>>>(feat_t, invn_t, fnt, kDT, (long)kNI * (kDT / 2));
  pack_w<<<256, BS, 0, stream>>>(W_v, Wp_v, kDV);
  pack_w<<<64, BS, 0, stream>>>(W_t, Wp_t, kDT);

  proj_wmma<kDV><<<dim3(kNI / 16, kE / 16), 32, 0, stream>>>(fnv, Wp_v, nrm_v, proj_v);
  proj_wmma<kDT><<<dim3(kNI / 16, kE / 16), 32, 0, stream>>>(fnt, Wp_t, nrm_t, proj_t);

  const size_t smemV = (size_t)16 * (kDV / 2) * 4 + 256 * 4;   // 129 KB (< 320 KB WGP LDS)
  const size_t smemT = (size_t)16 * (kDT / 2) * 4 + 256 * 4;   // 25.6 KB
  topk_sim<kDV><<<kNI / 16, 32, smemV, stream>>>(fnv, knn_v);
  topk_sim<kDT><<<kNI / 16, 32, smemT, stream>>>(fnt, knn_t);

  // ===== phase 2: hetero GCN, fusion, homo hops, loss =====
  zero_f32<<<GS, BS, 0, stream>>>(deg, kN);
  zero_f32<<<1, 32, 0, stream>>>(scal, 8);
  degree_kernel<<<(kNE + BS - 1) / BS, BS, 0, stream>>>(eu, ei, deg);

  // modality v
  copy_f32<<<GS, BS, 0, stream>>>(hA, pref_v, (long)kNU * kE);
  copy_f32<<<GS, BS, 0, stream>>>(hA + (size_t)kNU * kE, proj_v, (long)kNI * kE);
  copy_f32<<<GS, BS, 0, stream>>>(out_v, hA, nNE);
  zero_f32<<<GS, BS, 0, stream>>>(hB, nNE);
  gcn_scatter<<<scatterBlocks, BS, 0, stream>>>(hA, hB, eu, ei, deg);
  add_f32<<<GS, BS, 0, stream>>>(out_v, hB, nNE);
  zero_f32<<<GS, BS, 0, stream>>>(hA, nNE);
  gcn_scatter<<<scatterBlocks, BS, 0, stream>>>(hB, hA, eu, ei, deg);
  add_f32<<<GS, BS, 0, stream>>>(out_v, hA, nNE);

  // modality t (reuses hA/hB)
  copy_f32<<<GS, BS, 0, stream>>>(hA, pref_t, (long)kNU * kE);
  copy_f32<<<GS, BS, 0, stream>>>(hA + (size_t)kNU * kE, proj_t, (long)kNI * kE);
  copy_f32<<<GS, BS, 0, stream>>>(out_t, hA, nNE);
  zero_f32<<<GS, BS, 0, stream>>>(hB, nNE);
  gcn_scatter<<<scatterBlocks, BS, 0, stream>>>(hA, hB, eu, ei, deg);
  add_f32<<<GS, BS, 0, stream>>>(out_t, hB, nNE);
  zero_f32<<<GS, BS, 0, stream>>>(hA, nNE);
  gcn_scatter<<<scatterBlocks, BS, 0, stream>>>(hB, hA, eu, ei, deg);
  add_f32<<<GS, BS, 0, stream>>>(out_t, hA, nNE);

  combine_feats<<<(int)((nNE + BS - 1) / BS), BS, 0, stream>>>(out_v, out_t, alpha, u_f, i_f);

  item_homo<<<(int)(((long)kNI * kD2 + BS - 1) / BS), BS, 0, stream>>>(i_f, knn_v, knn_t, z_i);
  user_homo<<<kNU, kD2, 0, stream>>>(u_f, user_nb, nbw, z_u);

  bpr_kernel<<<(int)(((long)kB * 32 + BS - 1) / BS), BS, 0, stream>>>(z_u, z_i, u_ids, pos_ids,
                                                                      neg_ids, scal);
  sq_reduce<<<512, BS, 0, stream>>>(pref_v, (long)kNU * kE, scal + 1);
  sq_reduce<<<512, BS, 0, stream>>>(pref_t, (long)kNU * kE, scal + 1);
  finalize_kernel<<<1, 32, 0, stream>>>(scal, alpha, (float*)d_out);
}
